// VisualSemanticCrossAttn_62508954026408
// MI455X (gfx1250) — matrix operand order repaired and checked by reading, not compiled, not run on hardware
//
#include <hip/hip_runtime.h>
#include <hip/hip_bf16.h>

// Problem constants (from reference)
#define NQ   32768
#define CK   256
#define DIM  1024
#define NH   16
#define HD   64
#define DR   64
#define SCALE_F 0.125f   // 64^-0.5

typedef __attribute__((ext_vector_type(16))) __bf16 v16bf;
typedef __attribute__((ext_vector_type(8)))  __bf16 v8bf;
typedef __attribute__((ext_vector_type(4)))  __bf16 v4bf;
typedef __attribute__((ext_vector_type(8)))  float  v8f;
typedef __attribute__((ext_vector_type(4)))  int    v4i;

// Probe-confirmed (round 2): async-LDS + s_wait_asynccnt builtins available.
// Signature of global_load_async_to_lds_b128: 4 args, arg0 = addrspace(1) v4i*.
#if __has_builtin(__builtin_amdgcn_global_load_async_to_lds_b128)
#define ASYNC_LDS_COPY 1
#else
#define ASYNC_LDS_COPY 0
#endif

#if ASYNC_LDS_COPY
typedef __attribute__((address_space(1))) v4i* gv4i_p;   // global int4*
typedef __attribute__((address_space(3))) v4i* lv4i_p;   // LDS int4*
__device__ __forceinline__ void async_b128(const void* g, void* l) {
    __builtin_amdgcn_global_load_async_to_lds_b128((gv4i_p)g, (lv4i_p)l, 0, 0);
}
__device__ __forceinline__ void wait_async_all() {
#if __has_builtin(__builtin_amdgcn_s_wait_asynccnt)
    __builtin_amdgcn_s_wait_asynccnt(0);
#else
    asm volatile("s_wait_asynccnt 0" ::: "memory");
#endif
}
#endif

// Assemble a 16-element bf16 WMMA fragment from two 8-element (16B) runs.
// A-matrix: gap = 16 elements (runs at k and k+16 within the 32-wide K tile)
// B-matrix: gap = 16 elements too (same k order as A)
__device__ __forceinline__ v16bf frag2(const __bf16* p, int gap) {
    v8bf lo = *(const v8bf*)(p);
    v8bf hi = *(const v8bf*)(p + gap);
    return __builtin_shufflevector(lo, hi, 0,1,2,3,4,5,6,7,8,9,10,11,12,13,14,15);
}

// ---------------------------------------------------------------------------
// f32 -> bf16 conversion (vectorized, n must be a multiple of 4)
// ---------------------------------------------------------------------------
__global__ __launch_bounds__(256) void cvt_bf16(const float* __restrict__ in,
                                                __bf16* __restrict__ out, int n4) {
    int i = blockIdx.x * 256 + threadIdx.x;
    if (i >= n4) return;
    float4 v = ((const float4*)in)[i];
    v4bf o;
    o[0] = (__bf16)v.x; o[1] = (__bf16)v.y; o[2] = (__bf16)v.z; o[3] = (__bf16)v.w;
    ((v4bf*)out)[i] = o;
}

// ---------------------------------------------------------------------------
// b[n,h] = rope_bias[n,:] . Wrope[h,:]   (tiny: 33 MFLOP total)
// ---------------------------------------------------------------------------
__global__ __launch_bounds__(256) void rope_bias_kernel(const float* __restrict__ rb,
                                                        const float* __restrict__ Wr,
                                                        float* __restrict__ b) {
    int idx = blockIdx.x * 256 + threadIdx.x;     // 0 .. NQ*NH-1
    int n = idx >> 4, h = idx & (NH - 1);
    const float* r = rb + (size_t)n * DR;
    const float* w = Wr + (size_t)h * DR;
    float s = 0.f;
#pragma unroll
    for (int k = 0; k < DR; ++k) s += r[k] * w[k];
    b[idx] = s;
}

// ---------------------------------------------------------------------------
// V[C,D] -> Vt[D,C]  (bf16 transpose; 0.5 MB, negligible)
// ---------------------------------------------------------------------------
__global__ __launch_bounds__(256) void transpose_v(const __bf16* __restrict__ V,
                                                   __bf16* __restrict__ Vt) {
    int idx = blockIdx.x * 256 + threadIdx.x;     // 0 .. C*D-1
    int c = idx / DIM, d = idx % DIM;
    Vt[(size_t)d * CK + c] = V[idx];
}

// ---------------------------------------------------------------------------
// Tiled Linear GEMM: out[M,Nn] = A[M,K] @ W[Nn,K]^T  (bf16 in, f32 accum)
// 256 threads = 8 waves; block tile 128x128; K-step 64; wave tile 64x32.
// Double-buffered LDS pipeline:
//   - async path: GLOBAL_LOAD_ASYNC_TO_LDS_B128 for the next tile while
//     WMMAs consume the current one; s_wait_asynccnt 0 + one barrier / step.
//   - fallback:  prefetch next tile into VGPRs during compute, commit to the
//     other LDS buffer afterwards; one barrier / step.
// M % 128 == 0, Nn % 128 == 0, K % 64 == 0 (holds for all uses here).
// ---------------------------------------------------------------------------
#define LDB 80   // padded LDS row (elements); 160B rows keep 16B alignment
template <bool OUTF32>
__global__ __launch_bounds__(256) void gemm_bf16(const __bf16* __restrict__ A,
                                                 const __bf16* __restrict__ W,
                                                 void* __restrict__ outp,
                                                 int M, int Nn, int K) {
    __shared__ __align__(16) __bf16 As[2][128 * LDB];
    __shared__ __align__(16) __bf16 Bs[2][128 * LDB];
    const int t    = threadIdx.x;
    const int lane = t & 31, wv = t >> 5;
    const int wm = wv >> 2, wn = wv & 3;          // 2 x 4 wave grid over 128x128
    const int half = lane >> 4, l16 = lane & 15;
    const int m0 = blockIdx.y * 128, n0 = blockIdx.x * 128;

    // Staging chunk coordinates for this thread (4 x 16B per 128x64 tile)
    int rr[4], cc[4];
#pragma unroll
    for (int i = 0; i < 4; ++i) { int q = i * 256 + t; rr[i] = q >> 3; cc[i] = (q & 7) * 8; }

    v8f acc[4][2] = {};

    auto compute = [&](int buf) {
#pragma unroll
        for (int kk = 0; kk < 64; kk += 32) {
            v16bf af[4], bfr[2];
#pragma unroll
            for (int i = 0; i < 4; ++i)
                af[i] = frag2(&As[buf][(wm * 64 + i * 16 + l16) * LDB + kk + half * 8], 16);
#pragma unroll
            for (int j = 0; j < 2; ++j)
                bfr[j] = frag2(&Bs[buf][(wn * 32 + j * 16 + l16) * LDB + kk + half * 8], 16);
#pragma unroll
            for (int i = 0; i < 4; ++i)
#pragma unroll
                for (int j = 0; j < 2; ++j)
                    acc[i][j] = __builtin_amdgcn_wmma_f32_16x16x32_bf16(
                        false, af[i], false, bfr[j], (short)0, acc[i][j], false, false);
        }
    };

#if ASYNC_LDS_COPY
    // ---- async double-buffered pipeline ----
#pragma unroll
    for (int i = 0; i < 4; ++i) {
        async_b128(&A[(size_t)(m0 + rr[i]) * K + cc[i]], &As[0][rr[i] * LDB + cc[i]]);
        async_b128(&W[(size_t)(n0 + rr[i]) * K + cc[i]], &Bs[0][rr[i] * LDB + cc[i]]);
    }
    wait_async_all();
    __syncthreads();
    int buf = 0;
    for (int k0 = 0; k0 < K; k0 += 64) {
        if (k0 + 64 < K) {
#pragma unroll
            for (int i = 0; i < 4; ++i) {
                async_b128(&A[(size_t)(m0 + rr[i]) * K + k0 + 64 + cc[i]],
                           &As[buf ^ 1][rr[i] * LDB + cc[i]]);
                async_b128(&W[(size_t)(n0 + rr[i]) * K + k0 + 64 + cc[i]],
                           &Bs[buf ^ 1][rr[i] * LDB + cc[i]]);
            }
        }
        compute(buf);
        wait_async_all();
        __syncthreads();
        buf ^= 1;
    }
#else
    // ---- register-prefetch double-buffered pipeline ----
    uint4 ra[4], rb[4];
#pragma unroll
    for (int i = 0; i < 4; ++i) {
        ra[i] = *(const uint4*)&A[(size_t)(m0 + rr[i]) * K + cc[i]];
        rb[i] = *(const uint4*)&W[(size_t)(n0 + rr[i]) * K + cc[i]];
    }
#pragma unroll
    for (int i = 0; i < 4; ++i) {
        *(uint4*)&As[0][rr[i] * LDB + cc[i]] = ra[i];
        *(uint4*)&Bs[0][rr[i] * LDB + cc[i]] = rb[i];
    }
    __syncthreads();
    int buf = 0;
    for (int k0 = 0; k0 < K; k0 += 64) {
        const bool nxt = (k0 + 64) < K;
        if (nxt) {
#pragma unroll
            for (int i = 0; i < 4; ++i) {
                ra[i] = *(const uint4*)&A[(size_t)(m0 + rr[i]) * K + k0 + 64 + cc[i]];
                rb[i] = *(const uint4*)&W[(size_t)(n0 + rr[i]) * K + k0 + 64 + cc[i]];
            }
        }
        compute(buf);
        if (nxt) {
#pragma unroll
            for (int i = 0; i < 4; ++i) {
                *(uint4*)&As[buf ^ 1][rr[i] * LDB + cc[i]] = ra[i];
                *(uint4*)&Bs[buf ^ 1][rr[i] * LDB + cc[i]] = rb[i];
            }
        }
        __syncthreads();
        buf ^= 1;
    }
#endif

    // Epilogue: lane holds (m = r + 8*half, n = l16) within each 16x16 tile
#pragma unroll
    for (int i = 0; i < 4; ++i)
#pragma unroll
        for (int j = 0; j < 2; ++j) {
            int mb = m0 + wm * 64 + i * 16 + half * 8;
            int nb = n0 + wn * 32 + j * 16 + l16;
#pragma unroll
            for (int r = 0; r < 8; ++r) {
                size_t off = (size_t)(mb + r) * Nn + nb;
                if (OUTF32) ((float*)outp)[off]  = acc[i][j][r];
                else        ((__bf16*)outp)[off] = (__bf16)acc[i][j][r];
            }
        }
}

// ---------------------------------------------------------------------------
// Fused attention: one wave per (head, 16-query tile). 2 waves / block.
//   S[16,256] = (Qtile @ K_h^T)*scale + b      (WMMA, f32 -> LDS)
//   softmax over 256 (2 lanes per row, shfl_xor combine)
//   O[16,64]  = P~ @ V_h, rows scaled by 1/rowsum (WMMA)
// ---------------------------------------------------------------------------
#define SLD 264   // padded f32 row for S
#define PLD 272   // padded bf16 row for P (544B rows, 16B aligned)
__global__ __launch_bounds__(64) void attn_kernel(const __bf16* __restrict__ Qb,
                                                  const __bf16* __restrict__ Kb,
                                                  const __bf16* __restrict__ Vt,
                                                  const float* __restrict__ bias,
                                                  __bf16* __restrict__ AO) {
    __shared__ __align__(16) float  Ssm[2][16 * SLD];
    __shared__ __align__(16) __bf16 Psm[2][16 * PLD];
    __shared__ float Rs[2][16];

    const int lane = threadIdx.x & 31, wid = threadIdx.x >> 5;
    const int gw = blockIdx.x * 2 + wid;
    const int h  = gw & (NH - 1);
    const int q0 = (gw >> 4) * 16;
    const int half = lane >> 4, l16 = lane & 15;
    float*  S = Ssm[wid];
    __bf16* P = Psm[wid];

    // Q fragments for this 16-row tile, kept in registers for all 16 col-tiles
    v16bf aq[2];
#pragma unroll
    for (int kk = 0; kk < 2; ++kk)
        aq[kk] = frag2(Qb + (size_t)(q0 + l16) * DIM + h * HD + kk * 32 + half * 8, 16);

    float bm[8];
#pragma unroll
    for (int r = 0; r < 8; ++r)
        bm[r] = bias[(size_t)(q0 + half * 8 + r) * NH + h];

    // ---- logits: S = Q K^T * scale + b ----
    for (int ct = 0; ct < CK / 16; ++ct) {
        v8f acc = {};
#pragma unroll
        for (int kk = 0; kk < 2; ++kk) {
            v16bf bk = frag2(Kb + (size_t)(ct * 16 + l16) * DIM + h * HD + kk * 32 + half * 8, 16);
            acc = __builtin_amdgcn_wmma_f32_16x16x32_bf16(
                false, aq[kk], false, bk, (short)0, acc, false, false);
        }
#pragma unroll
        for (int r = 0; r < 8; ++r)
            S[(half * 8 + r) * SLD + ct * 16 + l16] = acc[r] * SCALE_F + bm[r];
    }
    __syncthreads();

    // ---- softmax over C=256 (lane pair {l, l^16} shares row l16) ----
    {
        float* sr = S + l16 * SLD + half * 128;
        float mx = -3.4e38f;
        for (int c = 0; c < 128; ++c) mx = fmaxf(mx, sr[c]);
        mx = fmaxf(mx, __shfl_xor(mx, 16));
        float sum = 0.f;
        __bf16* pr = P + l16 * PLD + half * 128;
        for (int c = 0; c < 128; ++c) {
            float e = __expf(sr[c] - mx);
            sum += e;
            pr[c] = (__bf16)e;            // unnormalized; 1/sum folded into output
        }
        sum += __shfl_xor(sum, 16);
        if (half == 0) Rs[wid][l16] = 1.0f / sum;
    }
    __syncthreads();

    // ---- O = P~ @ V_h (K = 256), scale rows by 1/sum ----
#pragma unroll
    for (int nt = 0; nt < 4; ++nt) {
        v8f acc = {};
        for (int kk = 0; kk < CK; kk += 32) {
            v16bf ap = frag2(&P[l16 * PLD + kk + half * 8], 16);
            v16bf bv = frag2(Vt + (size_t)(h * HD + nt * 16 + l16) * CK + kk + half * 8, 16);
            acc = __builtin_amdgcn_wmma_f32_16x16x32_bf16(
                false, ap, false, bv, (short)0, acc, false, false);
        }
#pragma unroll
        for (int r = 0; r < 8; ++r) {
            float sc = Rs[wid][half * 8 + r];
            AO[(size_t)(q0 + half * 8 + r) * DIM + h * HD + nt * 16 + l16] =
                (__bf16)(acc[r] * sc);
        }
    }
}

// ---------------------------------------------------------------------------
extern "C" void kernel_launch(void* const* d_in, const int* in_sizes, int n_in,
                              void* d_out, int out_size, void* d_ws, size_t ws_size,
                              hipStream_t stream) {
    const float* x    = (const float*)d_in[0];  // [N,D]
    const float* kv   = (const float*)d_in[1];  // [C,D]
    const float* rb   = (const float*)d_in[2];  // [N,64]
    const float* Wq   = (const float*)d_in[3];  // [D,D]
    const float* Wk   = (const float*)d_in[4];
    const float* Wv   = (const float*)d_in[5];
    const float* Wout = (const float*)d_in[6];
    const float* Wr   = (const float*)d_in[7];  // [H,64]
    float* out = (float*)d_out;                 // [N,D] f32

    // Workspace carve-up (~213 MB total)
    char* w = (char*)d_ws;
    auto carve = [&](size_t bytes) { void* p = w; w += (bytes + 255) & ~(size_t)255; return p; };
    __bf16* xb   = (__bf16*)carve((size_t)NQ * DIM * 2);
    __bf16* Qb   = (__bf16*)carve((size_t)NQ * DIM * 2);
    __bf16* AOb  = (__bf16*)carve((size_t)NQ * DIM * 2);
    __bf16* kvb  = (__bf16*)carve((size_t)CK * DIM * 2);
    __bf16* Kbm  = (__bf16*)carve((size_t)CK * DIM * 2);
    __bf16* Vbm  = (__bf16*)carve((size_t)CK * DIM * 2);
    __bf16* Vtb  = (__bf16*)carve((size_t)DIM * CK * 2);
    __bf16* Wqb  = (__bf16*)carve((size_t)DIM * DIM * 2);
    __bf16* Wkb  = (__bf16*)carve((size_t)DIM * DIM * 2);
    __bf16* Wvb  = (__bf16*)carve((size_t)DIM * DIM * 2);
    __bf16* Wob  = (__bf16*)carve((size_t)DIM * DIM * 2);
    float*  bb   = (float*)carve((size_t)NQ * NH * 4);

    // 1) conversions f32 -> bf16
    auto cvt = [&](const float* src, __bf16* dst, size_t n) {
        int n4 = (int)(n / 4);
        cvt_bf16<<<(n4 + 255) / 256, 256, 0, stream>>>(src, dst, n4);
    };
    cvt(x,    xb,  (size_t)NQ * DIM);
    cvt(kv,   kvb, (size_t)CK * DIM);
    cvt(Wq,   Wqb, (size_t)DIM * DIM);
    cvt(Wk,   Wkb, (size_t)DIM * DIM);
    cvt(Wv,   Wvb, (size_t)DIM * DIM);
    cvt(Wout, Wob, (size_t)DIM * DIM);

    // 2) rope bias  b[N,H]
    rope_bias_kernel<<<(NQ * NH) / 256, 256, 0, stream>>>(rb, Wr, bb);

    // 3) projections (WMMA GEMMs)
    gemm_bf16<false><<<dim3(DIM / 128, NQ / 128), 256, 0, stream>>>(xb,  Wqb, Qb,  NQ, DIM, DIM);
    gemm_bf16<false><<<dim3(DIM / 128, CK / 128), 256, 0, stream>>>(kvb, Wkb, Kbm, CK, DIM, DIM);
    gemm_bf16<false><<<dim3(DIM / 128, CK / 128), 256, 0, stream>>>(kvb, Wvb, Vbm, CK, DIM, DIM);

    // 4) V^T for contiguous B-fragments in P@V
    transpose_v<<<(CK * DIM) / 256, 256, 0, stream>>>(Vbm, Vtb);

    // 5) fused attention: (N/16)*H waves, 2 waves per block
    attn_kernel<<<(NQ / 16) * NH / 2, 64, 0, stream>>>(Qb, Kbm, Vtb, bb, AOb);

    // 6) output projection (f32 result straight to d_out)
    gemm_bf16<true><<<dim3(DIM / 128, NQ / 128), 256, 0, stream>>>(AOb, Wob, out, NQ, DIM, DIM);
}
